// Encoder_63015760167352
// MI455X (gfx1250) — matrix operand order repaired and checked by reading, not compiled, not run on hardware
//
#include <hip/hip_runtime.h>
#include <math.h>

// ---------------------------------------------------------------------------
// MI455X (gfx1250) fused ViT-decoupling block.
// Compute-bound (~445 GFLOP vs ~26MB live data, all L2-resident at 192MB L2):
// use bf16 WMMA (v_wmma_f32_16x16x32_bf16) for the three GEMMs, fuse the
// neighbor gather / GELU / per-neighbor softmax inside one workgroup per token
// so the 540MB "neighbor"/"edge" tensors never touch HBM.
// ---------------------------------------------------------------------------

typedef __attribute__((ext_vector_type(16))) __bf16 v16bf;
typedef __attribute__((ext_vector_type(8)))  __bf16 v8bf;
typedef __attribute__((ext_vector_type(8)))  float  v8f;

constexpr int Bb = 4, Vv = 8, Nn = 196, Dd = 768, Kk = 4;
constexpr int NB  = (Vv - 1) * Kk;        // 28 neighbors per token
constexpr int TOK = Bb * Vv * Nn;         // 6272 tokens
constexpr int LDP = Dd + 8;               // LDS row pitch (16B aligned, de-bank)

static __device__ __forceinline__ float gelu_exact(float x) {
  return 0.5f * x * (1.0f + erff(x * 0.7071067811865476f));
}

// A-fragment (16-bit 16x32, MxK): lane<16 holds K [k0..k0+7]+[k0+16..k0+23],
// lane>=16 holds K [k0+8..k0+15]+[k0+24..k0+31] of row M=lane&15.
static __device__ __forceinline__ v16bf frag_a(const __bf16* base, int stride, int k0) {
  const int lane = threadIdx.x & 31;
  const __bf16* p = base + (lane & 15) * stride + k0 + ((lane < 16) ? 0 : 8);
  v8bf lo = *(const v8bf*)p;
  v8bf hi = *(const v8bf*)(p + 16);
  v16bf f;
#pragma unroll
  for (int i = 0; i < 8; ++i) { f[i] = lo[i]; f[i + 8] = hi[i]; }
  return f;
}

// B-fragment (16-bit 32x16, KxN) from a pre-transposed [N][K] matrix:
// lane<16 holds K [k0..k0+7]+[k0+16..k0+23], lane>=16 holds K [k0+8..k0+15]+[k0+24..k0+31] of column N=lane&15.
static __device__ __forceinline__ v16bf frag_b(const __bf16* base, int stride, int k0) {
  const int lane = threadIdx.x & 31;
  const __bf16* p = base + (lane & 15) * stride + k0 + ((lane < 16) ? 0 : 8);
  v8bf lo = *(const v8bf*)p;
  v8bf hi = *(const v8bf*)(p + 16);
  v16bf f;
#pragma unroll
  for (int i = 0; i < 8; ++i) { f[i] = lo[i]; f[i + 8] = hi[i]; }
  return f;
}

static __device__ __forceinline__ v8f wmma_bf16(v16bf a, v16bf b, v8f c) {
  return __builtin_amdgcn_wmma_f32_16x16x32_bf16(false, a, false, b, (short)0, c,
                                                 false, false);
}

#define WSUM(s)                                                                \
  { s += __shfl_xor(s, 16); s += __shfl_xor(s, 8); s += __shfl_xor(s, 4);      \
    s += __shfl_xor(s, 2);  s += __shfl_xor(s, 1); }

#define ARGMIN_STEP(bv, bi_, m)                                                \
  { float ov = __shfl_xor(bv, m); int oi = __shfl_xor(bi_, m);                 \
    if (ov < bv || (ov == bv && oi < bi_)) { bv = ov; bi_ = oi; } }

// --- prep: W[K][N] fp32 -> W^T[N][K] bf16 -----------------------------------
__global__ void k_transpose_bf16(const float* __restrict__ src,
                                 __bf16* __restrict__ dst, int K, int Nc) {
  for (long i = (long)blockIdx.x * blockDim.x + threadIdx.x; i < (long)K * Nc;
       i += (long)gridDim.x * blockDim.x) {
    int n = (int)(i / K), k = (int)(i % K);
    dst[i] = (__bf16)src[(long)k * Nc + n];
  }
}

__global__ void k_f32_to_bf16(const float* __restrict__ src,
                              __bf16* __restrict__ dst, long n) {
  for (long i = (long)blockIdx.x * blockDim.x + threadIdx.x; i < n;
       i += (long)gridDim.x * blockDim.x)
    dst[i] = (__bf16)src[i];
}

// --- squared norms, one wave per token --------------------------------------
__global__ __launch_bounds__(256) void k_sqnorm(const float* __restrict__ x,
                                                float* __restrict__ sq) {
  int w = threadIdx.x >> 5, lane = threadIdx.x & 31;
  int row = blockIdx.x * 8 + w;
  if (row >= TOK) return;
  const float* p = x + (long)row * Dd;
  float s = 0.f;
#pragma unroll
  for (int i = 0; i < Dd / 32; ++i) { float v = p[lane + 32 * i]; s += v * v; }
  WSUM(s);
  if (lane == 0) sq[row] = s;
}

// --- KNN: one block per query token; 7 other views x top-4 ------------------
__global__ __launch_bounds__(256) void k_knn(const float* __restrict__ x,
                                             const float* __restrict__ sq,
                                             int* __restrict__ idxo) {
  __shared__ float xq[Dd];
  __shared__ float dist[Nn];
  int t = blockIdx.x, tid = threadIdx.x, w = tid >> 5, lane = tid & 31;
  int b_ = t / (Vv * Nn), v_ = (t / Nn) % Vv;
  for (int i = tid; i < Dd; i += 256) xq[i] = x[(long)t * Dd + i];
  __syncthreads();
  float sqq = sq[t];
  for (int vi = 0; vi < Vv - 1; ++vi) {
    int v1 = (vi < v_) ? vi : vi + 1;
    int cbase = (b_ * Vv + v1) * Nn;
    for (int c = w; c < Nn; c += 8) {         // one wave per candidate
      const float* xc = x + (long)(cbase + c) * Dd;
      float s = 0.f;
#pragma unroll
      for (int i = 0; i < Dd / 32; ++i) s += xq[lane + 32 * i] * xc[lane + 32 * i];
      WSUM(s);
      if (lane == 0) dist[c] = sqq + sq[cbase + c] - 2.f * s;
    }
    __syncthreads();
    if (w == 0) {                              // wave0: 4x masked argmin
      int sel[4];
#pragma unroll
      for (int si = 0; si < 4; ++si) {
        float bv = 3.4e38f; int bi_ = 0x7fffffff;
        for (int c = lane; c < Nn; c += 32) {
          float dv = dist[c];
          if (dv < bv) { bv = dv; bi_ = c; }
        }
        ARGMIN_STEP(bv, bi_, 16); ARGMIN_STEP(bv, bi_, 8);
        ARGMIN_STEP(bv, bi_, 4);  ARGMIN_STEP(bv, bi_, 2);
        ARGMIN_STEP(bv, bi_, 1);
        sel[si] = bi_;
        if (lane == 0) dist[bi_] = 3.4e38f;    // same-wave LDS: in order
      }
      if (lane == 0) {                         // ascending index order
#pragma unroll
        for (int a = 0; a < 3; ++a)
#pragma unroll
          for (int b2 = 0; b2 < 3; ++b2)
            if (sel[b2] > sel[b2 + 1]) { int tmp = sel[b2]; sel[b2] = sel[b2 + 1]; sel[b2 + 1] = tmp; }
#pragma unroll
        for (int q = 0; q < 4; ++q) idxo[t * NB + vi * Kk + q] = sel[q];
      }
    }
    __syncthreads();
  }
}

// --- fused per-token: edge GEMM + GELU, fusion GEMM + k-softmax + agg -------
__global__ __launch_bounds__(256) void k_token(const float* __restrict__ x,
                                               const float* __restrict__ b_edge,
                                               const __bf16* __restrict__ wTe,
                                               const __bf16* __restrict__ wTf,
                                               const int* __restrict__ idx,
                                               __bf16* __restrict__ aggbf) {
  __shared__ __align__(16) __bf16 Abuf[32 * LDP];   // neighbor - x (rows 28..31 zero)
  __shared__ __align__(16) __bf16 Ebuf[32 * LDP];   // edge = GELU(A We + b)
  __shared__ float xtok[Dd];
  int t = blockIdx.x, tid = threadIdx.x, w = tid >> 5, lane = tid & 31;
  int b_ = t / (Vv * Nn), v_ = (t / Nn) % Vv;
  for (int i = tid; i < Dd; i += 256) xtok[i] = x[(long)t * Dd + i];
  __syncthreads();
  for (int e = tid; e < NB * Dd; e += 256) {
    int j = e / Dd, c = e - j * Dd;
    int vs = j >> 2;
    int v1 = (vs < v_) ? vs : vs + 1;
    int n1 = idx[t * NB + j];
    long nb = (long)((b_ * Vv + v1) * Nn + n1) * Dd;
    Abuf[j * LDP + c] = (__bf16)(x[nb + c] - xtok[c]);
  }
  for (int e = tid; e < 4 * Dd; e += 256)
    Abuf[(28 + e / Dd) * LDP + (e % Dd)] = (__bf16)0.f;
  __syncthreads();

  // GEMM1: E = GELU(A @ We + b_edge). Each wave owns a 96-col strip.
  for (int g = 0; g < 6; ++g) {
    int colbase = w * 96 + g * 16;
    v8f acc0 = {0, 0, 0, 0, 0, 0, 0, 0}, acc1 = {0, 0, 0, 0, 0, 0, 0, 0};
    for (int kt = 0; kt < Dd / 32; ++kt) {
      int k0 = kt * 32;
      v16bf a0 = frag_a(Abuf, LDP, k0);
      v16bf a1 = frag_a(Abuf + 16 * LDP, LDP, k0);
      v16bf bb = frag_b(wTe + (long)colbase * Dd, Dd, k0);
      acc0 = wmma_bf16(a0, bb, acc0);
      acc1 = wmma_bf16(a1, bb, acc1);
    }
    int col = colbase + (lane & 15);
    float bias = b_edge[col];
    int mb = (lane < 16) ? 0 : 8;
#pragma unroll
    for (int r = 0; r < 8; ++r) {
      Ebuf[(mb + r) * LDP + col]      = (__bf16)gelu_exact(acc0[r] + bias);
      Ebuf[(16 + mb + r) * LDP + col] = (__bf16)gelu_exact(acc1[r] + bias);
    }
  }
  __syncthreads();

  // GEMM2: F = E @ Wf, then softmax over the 28 neighbors (M axis) per
  // column and agg = sum_k E*softmax(F). WMMA C layout: column 'col' lives in
  // lanes (col, col+16), 8 M-rows each per tile -> one shfl_xor(16) reduce.
  for (int g = 0; g < 6; ++g) {
    int colbase = w * 96 + g * 16;
    v8f acc0 = {0, 0, 0, 0, 0, 0, 0, 0}, acc1 = {0, 0, 0, 0, 0, 0, 0, 0};
    for (int kt = 0; kt < Dd / 32; ++kt) {
      int k0 = kt * 32;
      v16bf a0 = frag_a(Ebuf, LDP, k0);
      v16bf a1 = frag_a(Ebuf + 16 * LDP, LDP, k0);
      v16bf bb = frag_b(wTf + (long)colbase * Dd, Dd, k0);
      acc0 = wmma_bf16(a0, bb, acc0);
      acc1 = wmma_bf16(a1, bb, acc1);
    }
    int col = colbase + (lane & 15);
    int mb = (lane < 16) ? 0 : 8;
    int nvalid1 = (lane < 16) ? 8 : 4;     // tile1 rows 16..23 / 24..27 valid
    float mx = -3.4e38f;
#pragma unroll
    for (int r = 0; r < 8; ++r) mx = fmaxf(mx, acc0[r]);
#pragma unroll
    for (int r = 0; r < 8; ++r) if (r < nvalid1) mx = fmaxf(mx, acc1[r]);
    mx = fmaxf(mx, __shfl_xor(mx, 16));
    float p0[8], p1[8], s = 0.f;
#pragma unroll
    for (int r = 0; r < 8; ++r) { p0[r] = expf(acc0[r] - mx); s += p0[r]; }
#pragma unroll
    for (int r = 0; r < 8; ++r) {
      p1[r] = (r < nvalid1) ? expf(acc1[r] - mx) : 0.f;
      s += p1[r];
    }
    s += __shfl_xor(s, 16);
    float inv = 1.f / s;
    float agg = 0.f;
#pragma unroll
    for (int r = 0; r < 8; ++r) agg += (float)Ebuf[(mb + r) * LDP + col] * p0[r];
#pragma unroll
    for (int r = 0; r < 8; ++r)
      if (r < nvalid1) agg += (float)Ebuf[(16 + mb + r) * LDP + col] * p1[r];
    agg *= inv;
    agg += __shfl_xor(agg, 16);
    if (lane < 16) aggbf[(long)t * Dd + col] = (__bf16)agg;
  }
}

// --- out = x + [x, agg] @ Wo + b_off : 6272 x 768 x 1536 WMMA GEMM ----------
__global__ __launch_bounds__(256) void k_out(const __bf16* __restrict__ xbf,
                                             const __bf16* __restrict__ aggbf,
                                             const __bf16* __restrict__ wTo,
                                             const float* __restrict__ x,
                                             const float* __restrict__ b_off,
                                             float* __restrict__ out) {
  int w = threadIdx.x >> 5, lane = threadIdx.x & 31;
  int rb = blockIdx.y * 128 + w * 16;   // 49*128 == 6272 exactly
  int nb = blockIdx.x * 64;
  v8f acc[4];
#pragma unroll
  for (int j = 0; j < 4; ++j) acc[j] = (v8f){0, 0, 0, 0, 0, 0, 0, 0};
  for (int kt = 0; kt < 48; ++kt) {
    const __bf16* ab = (kt < 24) ? (xbf + (long)rb * Dd) : (aggbf + (long)rb * Dd);
    int k0 = (kt < 24) ? kt * 32 : (kt - 24) * 32;
    v16bf a = frag_a(ab, Dd, k0);
#pragma unroll
    for (int j = 0; j < 4; ++j) {
      v16bf bb = frag_b(wTo + (long)(nb + j * 16) * 1536, 1536, kt * 32);
      acc[j] = wmma_bf16(a, bb, acc[j]);
    }
  }
  int mb = (lane < 16) ? 0 : 8;
#pragma unroll
  for (int j = 0; j < 4; ++j) {
    int col = nb + j * 16 + (lane & 15);
    float bias = b_off[col];
#pragma unroll
    for (int r = 0; r < 8; ++r) {
      int row = rb + mb + r;
      out[(long)row * Dd + col] = x[(long)row * Dd + col] + bias + acc[j][r];
    }
  }
}

extern "C" void kernel_launch(void* const* d_in, const int* in_sizes, int n_in,
                              void* d_out, int out_size, void* d_ws, size_t ws_size,
                              hipStream_t stream) {
  (void)in_sizes; (void)n_in; (void)out_size; (void)ws_size;
  const float* x        = (const float*)d_in[0];
  const float* w_edge   = (const float*)d_in[1];
  const float* b_edge   = (const float*)d_in[2];
  const float* w_fusion = (const float*)d_in[3];
  const float* w_offset = (const float*)d_in[4];
  const float* b_offset = (const float*)d_in[5];

  char* p = (char*)d_ws;
  auto carve = [&](size_t bytes) { void* r = (void*)p; p += (bytes + 255) & ~(size_t)255; return r; };
  __bf16* wTe   = (__bf16*)carve((size_t)Dd * Dd * 2);          // We^T  [768][768]
  __bf16* wTf   = (__bf16*)carve((size_t)Dd * Dd * 2);          // Wf^T  [768][768]
  __bf16* wTo   = (__bf16*)carve((size_t)Dd * 1536 * 2);        // Wo^T  [768][1536]
  __bf16* xbf   = (__bf16*)carve((size_t)TOK * Dd * 2);         // x bf16
  __bf16* aggbf = (__bf16*)carve((size_t)TOK * Dd * 2);         // agg bf16
  float*  sq    = (float*)carve((size_t)TOK * 4);
  int*    idxw  = (int*)carve((size_t)TOK * NB * 4);

  k_transpose_bf16<<<(Dd * Dd + 255) / 256, 256, 0, stream>>>(w_edge, wTe, Dd, Dd);
  k_transpose_bf16<<<(Dd * Dd + 255) / 256, 256, 0, stream>>>(w_fusion, wTf, Dd, Dd);
  k_transpose_bf16<<<(1536 * Dd + 255) / 256, 256, 0, stream>>>(w_offset, wTo, 1536, Dd);
  k_f32_to_bf16<<<((long)TOK * Dd + 255) / 256, 256, 0, stream>>>(x, xbf, (long)TOK * Dd);
  k_sqnorm<<<TOK / 8, 256, 0, stream>>>(x, sq);
  k_knn<<<TOK, 256, 0, stream>>>(x, sq, idxw);
  k_token<<<TOK, 256, 0, stream>>>(x, b_edge, wTe, wTf, idxw, aggbf);
  k_out<<<dim3(Dd / 64, TOK / 128), 256, 0, stream>>>(xbf, aggbf, wTo, x, b_offset,
                                                      (float*)d_out);
}
